// MultiheadFlashDiff2_49254684950793
// MI455X (gfx1250) — matrix operand order repaired and checked by reading, not compiled, not run on hardware
//
#include <hip/hip_runtime.h>
#include <hip/hip_bf16.h>

typedef __attribute__((ext_vector_type(16))) __bf16 v16bf;
typedef __attribute__((ext_vector_type(8)))  __bf16 v8bf;
typedef __attribute__((ext_vector_type(8)))  float  v8f;

#define SEQ   2048
#define EMB   1024
#define NH    16
#define HD    32
#define LI            0.78360576656f      /* 0.8 - 0.6*exp(-3.6) */
#define ONE_MINUS_LI  0.21639423344f
#define QK_SCALE      0.17677669529663687f /* 32^-0.5 */

static __device__ __forceinline__ __bf16 f2bf(float f) {
  unsigned u = __builtin_bit_cast(unsigned, f);
  u += 0x7FFFu + ((u >> 16) & 1u);           // round-to-nearest-even
  return __builtin_bit_cast(__bf16, (unsigned short)(u >> 16));
}

static __device__ __forceinline__ v8f wmma_bf16(v16bf a, v16bf b, v8f c) {
  return __builtin_amdgcn_wmma_f32_16x16x32_bf16(false, a, false, b, (short)0, c, false, false);
}

// gfx1250 async memory->LDS copy (ASYNCcnt path); builtins don't cover these.
static __device__ __forceinline__ void async_load_b128(unsigned lds_off, const void* gaddr) {
  asm volatile("global_load_async_to_lds_b128 %0, %1, off"
               :: "v"(lds_off), "v"(gaddr) : "memory");
}

static __device__ __forceinline__ float red_max16(float v) {
#pragma unroll
  for (int m = 1; m < 16; m <<= 1) v = fmaxf(v, __shfl_xor(v, m, 32));
  return v;
}
static __device__ __forceinline__ float red_sum16(float v) {
#pragma unroll
  for (int m = 1; m < 16; m <<= 1) v += __shfl_xor(v, m, 32);
  return v;
}
static __device__ __forceinline__ float red_sum32(float v) {
#pragma unroll
  for (int m = 1; m < 32; m <<= 1) v += __shfl_xor(v, m, 32);
  return v;
}

// ---------------------------------------------------------------- casts
__global__ void cast_f32_to_bf16(const float* __restrict__ in,
                                 __bf16* __restrict__ out, int n) {
  int i = (blockIdx.x * blockDim.x + threadIdx.x) * 4;
  if (i + 3 < n) {
    float4 v = *(const float4*)(in + i);
    out[i]     = f2bf(v.x);
    out[i + 1] = f2bf(v.y);
    out[i + 2] = f2bf(v.z);
    out[i + 3] = f2bf(v.w);
  } else {
    for (; i < n; ++i) out[i] = f2bf(in[i]);
  }
}

// ---------------------------------------------------------------- RoPE
// Q/K fp32 (rows = b*t, 1024 ch = 32 heads x 32 dims, interleaved pairs)
__global__ void rope_cast(const float* __restrict__ Qf, const float* __restrict__ Kf,
                          __bf16* __restrict__ Qb, __bf16* __restrict__ Kb) {
  const int NP = 2 * SEQ * (EMB / 2);  // pairs per tensor
  int gid = blockIdx.x * blockDim.x + threadIdx.x;
  const float* src = Qf;
  __bf16* dst = Qb;
  int idx = gid;
  if (gid >= NP) { src = Kf; dst = Kb; idx = gid - NP; }
  int row  = idx >> 9;          // / 512 pairs per row
  int ch2  = idx & 511;
  int head = ch2 >> 4;          // 0..31
  int j    = ch2 & 15;          // freq index
  int t    = row & (SEQ - 1);
  float ang = __powf(10000.f, -(float)j * (1.f / 15.f));
  float f = (float)t * ang;
  float cf = __cosf(f), sf = __sinf(f);
  size_t off = (size_t)row * EMB + head * HD + 2 * j;
  float x1 = src[off], x2 = src[off + 1];
  dst[off]     = f2bf(x1 * cf - x2 * sf);
  dst[off + 1] = f2bf(x1 * sf + x2 * cf);
}

// ---------------------------------------------------------------- GEMM
// C[M,N] fp32 = A[M,K] bf16 * W[N,K]^T bf16   (both contiguous over K)
// Double-buffered: tiles stream memory->LDS via global_load_async_to_lds_b128
// while the previous tile's WMMAs execute. Last iteration peeled so the
// steady-state loop is branch-free.
__global__ __launch_bounds__(256)
void gemm_bf16_nt(const __bf16* __restrict__ A, const __bf16* __restrict__ W,
                  float* __restrict__ C, int M, int N, int K) {
  __shared__ __bf16 Alds[2][128][40];   // pad-40 keeps every row 16B aligned
  __shared__ __bf16 Blds[2][128][40];
  const int tid  = threadIdx.x;
  const int lane = tid & 31, wave = tid >> 5;
  const int lh   = lane & 15, half = lane >> 4;
  const int wm   = wave & 1,  wn   = wave >> 1;      // 2x4 wave grid
  const int mb   = blockIdx.x * 128, nb = blockIdx.y * 128;
  const int lr   = tid >> 1;          // 0..127 tile row
  const int lc   = (tid & 1) * 16;    // 0 or 16 within 32-wide k chunk

  v8f acc[4][2];
#pragma unroll
  for (int i = 0; i < 4; i++)
#pragma unroll
    for (int j = 0; j < 2; j++) acc[i][j] = {};

  const int KT = K >> 5;

  auto issue_tile = [&](int kt) {
    const int buf = kt & 1;
    const __bf16* ga = A + (size_t)(mb + lr) * K + kt * 32 + lc;
    const __bf16* gb = W + (size_t)(nb + lr) * K + kt * 32 + lc;
    unsigned la = (unsigned)(unsigned long long)&Alds[buf][lr][lc];
    unsigned lb = (unsigned)(unsigned long long)&Blds[buf][lr][lc];
    async_load_b128(la,      ga);
    async_load_b128(la + 16, ga + 8);
    async_load_b128(lb,      gb);
    async_load_b128(lb + 16, gb + 8);
  };

  auto compute_tile = [&](int buf) {
    v16bf bfr[2];
#pragma unroll
    for (int ni = 0; ni < 2; ++ni) {
      const __bf16* p = &Blds[buf][wn * 32 + ni * 16 + lh][half * 16];
      ((v8bf*)&bfr[ni])[0] = *(const v8bf*)p;
      ((v8bf*)&bfr[ni])[1] = *(const v8bf*)(p + 8);
    }
#pragma unroll
    for (int mi = 0; mi < 4; ++mi) {
      const __bf16* p = &Alds[buf][wm * 64 + mi * 16 + lh][0];
      v16bf afr;
      ((v8bf*)&afr)[0] = *(const v8bf*)(p + half * 16);
      ((v8bf*)&afr)[1] = *(const v8bf*)(p + half * 16 + 8);
      acc[mi][0] = wmma_bf16(afr, bfr[0], acc[mi][0]);
      acc[mi][1] = wmma_bf16(afr, bfr[1], acc[mi][1]);
    }
  };

  issue_tile(0);
#pragma unroll 2
  for (int kt = 0; kt < KT - 1; ++kt) {
    issue_tile(kt + 1);                               // stream next tile
    asm volatile("s_wait_asynccnt 4" ::: "memory");   // tile kt's 4 ops done
    __syncthreads();
    compute_tile(kt & 1);
    __syncthreads();
  }
  asm volatile("s_wait_asynccnt 0" ::: "memory");     // final tile
  __syncthreads();
  compute_tile((KT - 1) & 1);

#pragma unroll
  for (int mi = 0; mi < 4; ++mi)
#pragma unroll
    for (int ni = 0; ni < 2; ++ni)
#pragma unroll
      for (int r = 0; r < 8; ++r) {
        int row = mb + wm * 64 + mi * 16 + r + 8 * half;
        int col = nb + wn * 32 + ni * 16 + lh;
        C[(size_t)row * N + col] = acc[mi][ni][r];
      }
}

// ---------------------------------------------------------------- attention
// grid.x = b(2)*h(16)*comp(2) = 64 ; grid.y = SEQ/128 ; 8 waves x 16 q-rows
__global__ __launch_bounds__(256)
void diff_attn_fa(const __bf16* __restrict__ Qb, const __bf16* __restrict__ Kb,
                  const __bf16* __restrict__ Vb, float* __restrict__ Oc) {
  __shared__ __bf16 Vlds[64][40];       // V chunk transposed: [e][kk]
  __shared__ __bf16 Plds[8][16][40];    // per-wave P staging
  const int tid  = threadIdx.x, lane = tid & 31, wave = tid >> 5;
  const int lh   = lane & 15, half = lane >> 4;
  const int bh   = blockIdx.x;
  const int b = bh >> 5, h = (bh >> 1) & 15, c = bh & 1;
  const int t0 = blockIdx.y * 128 + wave * 16;

  // Q fragment (A-layout): row = t0+lh, K split per lane-half
  const __bf16* qrow = Qb + (size_t)(b * SEQ + t0 + lh) * EMB + h * 64 + c * 32;
  v16bf qf;
  ((v8bf*)&qf)[0] = *(const v8bf*)(qrow + half * 16);
  ((v8bf*)&qf)[1] = *(const v8bf*)(qrow + half * 16 + 8);

  float mrow[8], lrow[8];
  v8f o[4];
#pragma unroll
  for (int r = 0; r < 8; r++) { mrow[r] = -1e30f; lrow[r] = 0.f; }
#pragma unroll
  for (int e = 0; e < 4; e++) o[e] = {};
  const v8f zc = {};

  const __bf16* vbase = Vb + (size_t)(b * SEQ) * EMB + h * 64;

  for (int s0 = 0; s0 < SEQ; s0 += 32) {
    { // stage V chunk (32 keys x 64 dims) transposed into LDS
      int kk = tid >> 3;
      int e8 = (tid & 7) * 8;
      v8bf vv = *(const v8bf*)(vbase + (size_t)(s0 + kk) * EMB + e8);
#pragma unroll
      for (int j = 0; j < 8; j++) Vlds[e8 + j][kk] = vv[j];
    }
    __syncthreads();

    // K fragments (B-layout: col = key = lh, K-dim contiguous per lane-half)
    v16bf kf0, kf1;
    {
      const __bf16* kr0 = Kb + (size_t)(b * SEQ + s0 + lh) * EMB + h * 64 + c * 32;
      ((v8bf*)&kf0)[0] = *(const v8bf*)(kr0 + half * 16);
      ((v8bf*)&kf0)[1] = *(const v8bf*)(kr0 + half * 16 + 8);
      const __bf16* kr1 = Kb + (size_t)(b * SEQ + s0 + 16 + lh) * EMB + h * 64 + c * 32;
      ((v8bf*)&kf1)[0] = *(const v8bf*)(kr1 + half * 16);
      ((v8bf*)&kf1)[1] = *(const v8bf*)(kr1 + half * 16 + 8);
    }
    v8f sv0 = wmma_bf16(qf, kf0, zc);
    v8f sv1 = wmma_bf16(qf, kf1, zc);

    // online softmax (rows live in 16-lane halves; VGPR r = row r + 8*half)
    float p0[8], p1[8];
#pragma unroll
    for (int r = 0; r < 8; r++) {
      float a  = sv0[r] * QK_SCALE;
      float bb = sv1[r] * QK_SCALE;
      float rm = red_max16(fmaxf(a, bb));
      float mn = fmaxf(mrow[r], rm);
      float corr = __expf(mrow[r] - mn);
      float e0 = __expf(a - mn), e1 = __expf(bb - mn);
      lrow[r] = lrow[r] * corr + red_sum16(e0 + e1);
      mrow[r] = mn;
      p0[r] = e0;
      p1[r] = e1;
#pragma unroll
      for (int e = 0; e < 4; e++) o[e][r] *= corr;
    }

    // stage P (C-layout -> LDS), reload as A-fragment
#pragma unroll
    for (int r = 0; r < 8; r++) {
      Plds[wave][r + 8 * half][lh]      = f2bf(p0[r]);
      Plds[wave][r + 8 * half][16 + lh] = f2bf(p1[r]);
    }
    v16bf pf;
    ((v8bf*)&pf)[0] = *(const v8bf*)(&Plds[wave][lh][half * 16]);
    ((v8bf*)&pf)[1] = *(const v8bf*)(&Plds[wave][lh][half * 16 + 8]);

#pragma unroll
    for (int e = 0; e < 4; e++) {
      v16bf vf;
      const __bf16* p = &Vlds[e * 16 + lh][half * 16];
      ((v8bf*)&vf)[0] = *(const v8bf*)p;
      ((v8bf*)&vf)[1] = *(const v8bf*)(p + 8);
      o[e] = wmma_bf16(pf, vf, o[e]);
    }
    __syncthreads();
  }

  // epilogue: O /= l ; write fp32 to (b,t, h*128 + c*64 + e)
#pragma unroll
  for (int r = 0; r < 8; r++) {
    float inv = 1.f / lrow[r];
    int t = t0 + r + 8 * half;
    size_t base = (size_t)(b * SEQ + t) * 2048 + h * 128 + c * 64;
#pragma unroll
    for (int e = 0; e < 4; e++) Oc[base + e * 16 + lh] = o[e][r] * inv;
  }
}

// ---------------------------------------------------------------- combine + RMS
__global__ void combine_rms(const float* __restrict__ attn,
                            const float* __restrict__ lq1, const float* __restrict__ lk1,
                            const float* __restrict__ lq2, const float* __restrict__ lk2,
                            const float* __restrict__ rmsw,
                            __bf16* __restrict__ A2) {
  int lane = threadIdx.x & 31, wave = threadIdx.x >> 5;
  int row = blockIdx.x * 8 + wave;          // (b*t)*16 + h, in [0, 65536)
  float s1 = 0.f, s2 = 0.f;
#pragma unroll
  for (int i = 0; i < HD; i++) { s1 += lq1[i] * lk1[i]; s2 += lq2[i] * lk2[i]; }
  float lam = __expf(s1) - __expf(s2) + LI;

  int bt = row >> 4, h = row & 15;
  size_t base = (size_t)bt * 2048 + h * 128;
  int e = lane * 2;
  float2 a1 = *(const float2*)(attn + base + e);
  float2 a2 = *(const float2*)(attn + base + 64 + e);
  float ax = a1.x - lam * a2.x;
  float ay = a1.y - lam * a2.y;
  float ss = red_sum32(ax * ax + ay * ay);
  float rn = rsqrtf(ss * (1.f / 64.f) + 1e-5f);
  size_t ob = (size_t)bt * EMB + h * 64 + e;
  A2[ob]     = f2bf(ax * rn * rmsw[e]     * ONE_MINUS_LI);
  A2[ob + 1] = f2bf(ay * rn * rmsw[e + 1] * ONE_MINUS_LI);
}

// ---------------------------------------------------------------- launch
extern "C" void kernel_launch(void* const* d_in, const int* in_sizes, int n_in,
                              void* d_out, int out_size, void* d_ws, size_t ws_size,
                              hipStream_t stream) {
  const float* x    = (const float*)d_in[0];
  const float* Wq   = (const float*)d_in[1];
  const float* Wk   = (const float*)d_in[2];
  const float* Wv   = (const float*)d_in[3];
  const float* Wo   = (const float*)d_in[4];
  const float* lq1  = (const float*)d_in[5];
  const float* lk1  = (const float*)d_in[6];
  const float* lq2  = (const float*)d_in[7];
  const float* lk2  = (const float*)d_in[8];
  const float* rmsw = (const float*)d_in[9];
  float* out = (float*)d_out;
  (void)in_sizes; (void)n_in; (void)out_size; (void)ws_size;

  const int M = 2 * SEQ;  // 4096 rows
  char* w = (char*)d_ws;
  size_t off = 0;
  auto alloc = [&](size_t bytes) -> void* {
    void* p = w + off;
    off += (bytes + 255) & ~(size_t)255;
    return p;
  };
  __bf16* Wq_b = (__bf16*)alloc((size_t)EMB * EMB * 2);
  __bf16* Wk_b = (__bf16*)alloc((size_t)EMB * EMB * 2);
  __bf16* Wv_b = (__bf16*)alloc((size_t)EMB * EMB * 2);
  __bf16* Wo_b = (__bf16*)alloc((size_t)EMB * EMB * 2);
  __bf16* x_b  = (__bf16*)alloc((size_t)M * EMB * 2);
  float*  Qf   = (float*)alloc((size_t)M * EMB * 4);
  float*  Kf   = (float*)alloc((size_t)M * EMB * 4);   // contiguous after Qf
  float*  Vf   = (float*)alloc((size_t)M * EMB * 4);
  __bf16* Qb   = (__bf16*)alloc((size_t)M * EMB * 2);
  __bf16* Kb   = (__bf16*)alloc((size_t)M * EMB * 2);
  __bf16* Vb   = (__bf16*)alloc((size_t)M * EMB * 2);
  float*  attn = Qf;            // reuse Qf+Kf (32MB) after RoPE consumed them
  __bf16* A2   = (__bf16*)Vf;   // reuse Vf after V cast consumed it

  const int nW = EMB * EMB;
  cast_f32_to_bf16<<<(nW / 4 + 255) / 256, 256, 0, stream>>>(Wq, Wq_b, nW);
  cast_f32_to_bf16<<<(nW / 4 + 255) / 256, 256, 0, stream>>>(Wk, Wk_b, nW);
  cast_f32_to_bf16<<<(nW / 4 + 255) / 256, 256, 0, stream>>>(Wv, Wv_b, nW);
  cast_f32_to_bf16<<<(nW / 4 + 255) / 256, 256, 0, stream>>>(Wo, Wo_b, nW);
  const int nX = M * EMB;
  cast_f32_to_bf16<<<(nX / 4 + 255) / 256, 256, 0, stream>>>(x, x_b, nX);

  dim3 gg(M / 128, EMB / 128);
  gemm_bf16_nt<<<gg, 256, 0, stream>>>(x_b, Wq_b, Qf, M, EMB, EMB);
  gemm_bf16_nt<<<gg, 256, 0, stream>>>(x_b, Wk_b, Kf, M, EMB, EMB);
  gemm_bf16_nt<<<gg, 256, 0, stream>>>(x_b, Wv_b, Vf, M, EMB, EMB);

  rope_cast<<<(2 * (M * EMB / 2)) / 256, 256, 0, stream>>>(Qf, Kf, Qb, Kb);
  cast_f32_to_bf16<<<(nX / 4 + 255) / 256, 256, 0, stream>>>(Vf, Vb, nX);

  diff_attn_fa<<<dim3(64, SEQ / 128), 256, 0, stream>>>(Qb, Kb, Vb, attn);

  combine_rms<<<(M * NH) / 8, 256, 0, stream>>>(attn, lq1, lk1, lq2, lk2, rmsw, A2);

  gemm_bf16_nt<<<gg, 256, 0, stream>>>(A2, Wo_b, out, M, EMB, EMB);
}
